// PointTransformerLayer_17214228922881
// MI455X (gfx1250) — matrix operand order repaired and checked by reading, not compiled, not run on hardware
//
#include <hip/hip_runtime.h>
#include <math.h>

typedef __attribute__((ext_vector_type(16))) _Float16 v16h;
typedef __attribute__((ext_vector_type(8)))  float    v8f;

#define ROWS   16384      // B*S*N = 1*4*4096
#define NPTS   4096
#define NSLICE 4
#define HID    64
#define KNN    16

// ---------------------------------------------------------------------------
// GELU (tanh approximation, matches jax.nn.gelu default approximate=True)
// ---------------------------------------------------------------------------
__device__ __forceinline__ float gelu_tanh(float v) {
    const float k0 = 0.7978845608028654f;           // sqrt(2/pi)
    float u = k0 * (v + 0.044715f * v * v * v);
    return 0.5f * v * (1.0f + tanhf(u));
}

// ---------------------------------------------------------------------------
// Kernel 0: pre-swizzle Wq/Wk/Wv/Wo (f32 [64,64] row-major, W[k][n]) into
// f16 WMMA B-fragments. Layout: [mtx][t (ntile)][u (ktile)][lane][e].
// ISA B layout (16-bit, 32x16): e<8 -> K=u*32+g*8+e, e>=8 -> K=u*32+16+g*8+(e-8)
// (g = lane>>4) within the 32-K tile; N = t*16 + (lane&15).
// ---------------------------------------------------------------------------
__global__ __launch_bounds__(256) void prep_wfrag(
    const float* __restrict__ Wq, const float* __restrict__ Wk,
    const float* __restrict__ Wv, const float* __restrict__ Wo,
    _Float16* __restrict__ wfrag)
{
    for (int idx = threadIdx.x; idx < 4 * 4 * 2 * 32 * 16; idx += 256) {
        int e    = idx & 15;
        int lane = (idx >> 4) & 31;
        int u    = (idx >> 9) & 1;
        int t    = (idx >> 10) & 3;
        int mtx  = idx >> 12;
        const float* W = (mtx == 0) ? Wq : (mtx == 1) ? Wk : (mtx == 2) ? Wv : Wo;
        int g = lane >> 4, nl = lane & 15;
        int K = u * 32 + g * 8 + (e & 7) + ((e >> 3) << 4);
        int N = t * 16 + nl;
        wfrag[idx] = (_Float16)W[K * 64 + N];
    }
}

// ---------------------------------------------------------------------------
// Kernel 1: KNN. One block = 256 query points of one slice; full slice pos
// (4096x3 f32 = 48KB) staged in LDS. Scan all candidates in lockstep
// (broadcast LDS reads), maintain sorted top-16 in registers.
// Stable strict '<' matches jax.lax.top_k tie-breaking (lower index first).
// ---------------------------------------------------------------------------
__global__ __launch_bounds__(256) void knn_kernel(
    const float* __restrict__ pos, int* __restrict__ nbr)
{
    __shared__ float sp[NPTS * 3];
    const int s = blockIdx.y;
    const float* p = pos + (size_t)s * NPTS * 3;
    for (int i = threadIdx.x; i < NPTS * 3; i += 256) sp[i] = p[i];
    __syncthreads();

    const int nn = blockIdx.x * 256 + threadIdx.x;
    const float px = sp[nn * 3 + 0];
    const float py = sp[nn * 3 + 1];
    const float pz = sp[nn * 3 + 2];

    float bd[KNN];
    int   bi[KNN];
#pragma unroll
    for (int t = 0; t < KNN; ++t) { bd[t] = 3.0e38f; bi[t] = 0; }

    for (int m = 0; m < NPTS; ++m) {
        float dx = sp[m * 3 + 0] - px;
        float dy = sp[m * 3 + 1] - py;
        float dz = sp[m * 3 + 2] - pz;
        float d  = dx * dx + dy * dy + dz * dz;
        if (d < bd[KNN - 1]) {
            float cd = d; int ci = m;
#pragma unroll
            for (int t = 0; t < KNN; ++t) {
                if (cd < bd[t]) {
                    float tf = bd[t]; bd[t] = cd; cd = tf;
                    int   ti = bi[t]; bi[t] = ci; ci = ti;
                }
            }
        }
    }

    const int n = s * NPTS + nn;
#pragma unroll
    for (int t = 0; t < KNN; ++t) nbr[n * KNN + t] = bi[t];
}

// ---------------------------------------------------------------------------
// Kernel 2: fused q/k/v projection via WMMA f16 + pos_enc epilogue.
// Per wave: one 16-row tile. Outputs (f32):
//   qw  = (x@Wq + bq + pe) * Wa / sqrt(H)
//   kpe = (x@Wk + bk) + pe
//   vv  =  x@Wv + bv
// where pe = pos@Wpe + bpe.
// A-fragment ISA layout (f16 16x32): lane l, g=l>>4: elements 0..7 -> K =
// u*32+g*8+e, elements 8..15 -> K = u*32+16+g*8+(e-8).
// ---------------------------------------------------------------------------
__global__ __launch_bounds__(256) void proj_kernel(
    const float* __restrict__ x,   const float* __restrict__ pos,
    const float* __restrict__ bq,  const float* __restrict__ bk,
    const float* __restrict__ bv,  const float* __restrict__ Wpe,
    const float* __restrict__ bpe, const float* __restrict__ Wa,
    const _Float16* __restrict__ wfrag,
    float* __restrict__ qw, float* __restrict__ kpe, float* __restrict__ vv)
{
    const int lane    = threadIdx.x & 31;
    const int wave    = threadIdx.x >> 5;
    const int rowBase = blockIdx.x * 128 + wave * 16;
    const int g  = lane >> 4;
    const int lm = lane & 15;
    const int mrow = rowBase + lm;

    // A fragments (two K-tiles of 32)
    v16h a[2];
#pragma unroll
    for (int u = 0; u < 2; ++u) {
        const float* src = x + (size_t)mrow * 64 + u * 32 + g * 8;
#pragma unroll
        for (int e = 0; e < 8; ++e) a[u][e] = (_Float16)src[e];
#pragma unroll
        for (int e = 0; e < 8; ++e) a[u][8 + e] = (_Float16)src[16 + e];
    }

    v8f zero = {};
    v8f acc[3][4];
#pragma unroll
    for (int m = 0; m < 3; ++m)
#pragma unroll
        for (int t = 0; t < 4; ++t) acc[m][t] = zero;

#pragma unroll
    for (int mtx = 0; mtx < 3; ++mtx) {
#pragma unroll
        for (int t = 0; t < 4; ++t) {
#pragma unroll
            for (int u = 0; u < 2; ++u) {
                v16h b = *(const v16h*)(wfrag + ((((mtx * 4 + t) * 2 + u) * 32 + lane) << 4));
                acc[mtx][t] = __builtin_amdgcn_wmma_f32_16x16x32_f16(
                    false, a[u], false, b, (short)0, acc[mtx][t], false, false);
            }
        }
    }

    // Epilogue: pos_enc + biases + Wa/scale fold. C layout: reg r, lane l ->
    // M = r + (l>>4)*8, N = l&15 (within the 16-wide n-tile t).
#pragma unroll
    for (int t = 0; t < 4; ++t) {
        const int h = t * 16 + lm;
        const float wpe0 = Wpe[0 * 64 + h], wpe1 = Wpe[1 * 64 + h], wpe2 = Wpe[2 * 64 + h];
        const float bpeh = bpe[h], bqh = bq[h], bkh = bk[h], bvh = bv[h];
        const float wah  = Wa[h] * 0.125f;     // 1/sqrt(64)
#pragma unroll
        for (int r = 0; r < 8; ++r) {
            const int row = rowBase + g * 8 + r;
            const float pe = bpeh + pos[row * 3 + 0] * wpe0
                                  + pos[row * 3 + 1] * wpe1
                                  + pos[row * 3 + 2] * wpe2;
            const float qv = acc[0][t][r] + bqh + pe;
            qw [(size_t)row * 64 + h] = qv * wah;
            kpe[(size_t)row * 64 + h] = acc[1][t][r] + bkh + pe;
            vv [(size_t)row * 64 + h] = acc[2][t][r] + bvh;
        }
    }
}

// ---------------------------------------------------------------------------
// Kernel 3: attention. One thread per point.
// logit[j] = qw . kpe[idx_j] + pos_rel . (Wpd . qw) + (qw . bpd + ba)
// Softmax over 16, out = sum attn_j * v[idx_j].  All data L2-resident.
// ---------------------------------------------------------------------------
__global__ __launch_bounds__(256) void attn_kernel(
    const float* __restrict__ qw, const float* __restrict__ kpe,
    const float* __restrict__ vv, const int* __restrict__ nbr,
    const float* __restrict__ pos, const float* __restrict__ Wpd,
    const float* __restrict__ bpd, const float* __restrict__ ba,
    float* __restrict__ ao)
{
    const int n  = blockIdx.x * 256 + threadIdx.x;   // 0..16383
    const int s  = n >> 12;
    const int nn = n & 4095;

    // own query row
    float4 q4[16];
    const float4* qp = (const float4*)(qw + (size_t)n * 64);
#pragma unroll
    for (int i = 0; i < 16; ++i) q4[i] = qp[i];

    // wq3 = Wpd . qw  (3 values), c = qw . bpd + ba
    float w0 = 0.f, w1 = 0.f, w2 = 0.f, c = 0.f;
#pragma unroll
    for (int i = 0; i < 16; ++i) {
        const float4 q = q4[i];
        const int h = i * 4;
        w0 += Wpd[h] * q.x + Wpd[h + 1] * q.y + Wpd[h + 2] * q.z + Wpd[h + 3] * q.w;
        w1 += Wpd[64 + h] * q.x + Wpd[65 + h] * q.y + Wpd[66 + h] * q.z + Wpd[67 + h] * q.w;
        w2 += Wpd[128 + h] * q.x + Wpd[129 + h] * q.y + Wpd[130 + h] * q.z + Wpd[131 + h] * q.w;
        c  += bpd[h] * q.x + bpd[h + 1] * q.y + bpd[h + 2] * q.z + bpd[h + 3] * q.w;
    }
    c += ba[0];

    const int   selfRow = s * NPTS + nn;
    const float px = pos[selfRow * 3 + 0];
    const float py = pos[selfRow * 3 + 1];
    const float pz = pos[selfRow * 3 + 2];

    int   rows[KNN];
    float lg[KNN];
#pragma unroll
    for (int j = 0; j < KNN; ++j) {
        const int idx = nbr[n * KNN + j];
        const int row = s * NPTS + idx;
        rows[j] = row;
        const float4* kp = (const float4*)(kpe + (size_t)row * 64);
        float dsum = 0.f;
#pragma unroll
        for (int i = 0; i < 16; ++i) {
            const float4 k = kp[i];
            const float4 q = q4[i];
            dsum += q.x * k.x + q.y * k.y + q.z * k.z + q.w * k.w;
        }
        const float prx = pos[row * 3 + 0] - px;
        const float pry = pos[row * 3 + 1] - py;
        const float prz = pos[row * 3 + 2] - pz;
        lg[j] = dsum + prx * w0 + pry * w1 + prz * w2 + c;
    }

    // softmax over K=16
    float mx = lg[0];
#pragma unroll
    for (int j = 1; j < KNN; ++j) mx = fmaxf(mx, lg[j]);
    float se = 0.f;
#pragma unroll
    for (int j = 0; j < KNN; ++j) { lg[j] = __expf(lg[j] - mx); se += lg[j]; }
    const float inv = 1.0f / se;

    // weighted sum of neighbor V
    float4 acc[16];
#pragma unroll
    for (int i = 0; i < 16; ++i) acc[i] = make_float4(0.f, 0.f, 0.f, 0.f);
#pragma unroll
    for (int j = 0; j < KNN; ++j) {
        const float a = lg[j] * inv;
        const float4* vp = (const float4*)(vv + (size_t)rows[j] * 64);
#pragma unroll
        for (int i = 0; i < 16; ++i) {
            const float4 v = vp[i];
            acc[i].x += a * v.x; acc[i].y += a * v.y;
            acc[i].z += a * v.z; acc[i].w += a * v.w;
        }
    }

    float4* op = (float4*)(ao + (size_t)n * 64);
#pragma unroll
    for (int i = 0; i < 16; ++i) op[i] = acc[i];
}

// ---------------------------------------------------------------------------
// Kernel 4: output projection via WMMA + GELU + residual.
// out = x + gelu(ao @ Wo + bo)
// ---------------------------------------------------------------------------
__global__ __launch_bounds__(256) void out_kernel(
    const float* __restrict__ ao, const float* __restrict__ x,
    const float* __restrict__ bo, const _Float16* __restrict__ wfrag,
    float* __restrict__ out)
{
    const int lane    = threadIdx.x & 31;
    const int wave    = threadIdx.x >> 5;
    const int rowBase = blockIdx.x * 128 + wave * 16;
    const int g  = lane >> 4;
    const int lm = lane & 15;
    const int mrow = rowBase + lm;

    v16h a[2];
#pragma unroll
    for (int u = 0; u < 2; ++u) {
        const float* src = ao + (size_t)mrow * 64 + u * 32 + g * 8;
#pragma unroll
        for (int e = 0; e < 8; ++e) a[u][e] = (_Float16)src[e];
#pragma unroll
        for (int e = 0; e < 8; ++e) a[u][8 + e] = (_Float16)src[16 + e];
    }

    v8f zero = {};
    v8f acc[4];
#pragma unroll
    for (int t = 0; t < 4; ++t) acc[t] = zero;

#pragma unroll
    for (int t = 0; t < 4; ++t) {
#pragma unroll
        for (int u = 0; u < 2; ++u) {
            v16h b = *(const v16h*)(wfrag + ((((3 * 4 + t) * 2 + u) * 32 + lane) << 4));
            acc[t] = __builtin_amdgcn_wmma_f32_16x16x32_f16(
                false, a[u], false, b, (short)0, acc[t], false, false);
        }
    }

#pragma unroll
    for (int t = 0; t < 4; ++t) {
        const int h = t * 16 + lm;
        const float boh = bo[h];
#pragma unroll
        for (int r = 0; r < 8; ++r) {
            const int row = rowBase + g * 8 + r;
            const float val = acc[t][r] + boh;
            out[(size_t)row * 64 + h] = x[(size_t)row * 64 + h] + gelu_tanh(val);
        }
    }
}

// ---------------------------------------------------------------------------
// Launch
// ---------------------------------------------------------------------------
extern "C" void kernel_launch(void* const* d_in, const int* in_sizes, int n_in,
                              void* d_out, int out_size, void* d_ws, size_t ws_size,
                              hipStream_t stream) {
    const float* x   = (const float*)d_in[0];
    const float* pos = (const float*)d_in[1];
    const float* Wq  = (const float*)d_in[2];
    const float* bq  = (const float*)d_in[3];
    const float* Wk  = (const float*)d_in[4];
    const float* bk  = (const float*)d_in[5];
    const float* Wv  = (const float*)d_in[6];
    const float* bv  = (const float*)d_in[7];
    const float* Wpe = (const float*)d_in[8];
    const float* bpe = (const float*)d_in[9];
    const float* Wpd = (const float*)d_in[10];
    const float* bpd = (const float*)d_in[11];
    const float* Wa  = (const float*)d_in[12];
    const float* ba  = (const float*)d_in[13];
    const float* Wo  = (const float*)d_in[14];
    const float* bo  = (const float*)d_in[15];
    float* out = (float*)d_out;

    // workspace layout (all offsets 256B-aligned)
    float* QW  = (float*)d_ws;                    // 16384*64 f32 = 4 MB
    float* KPE = QW  + (size_t)ROWS * HID;        // 4 MB
    float* VV  = KPE + (size_t)ROWS * HID;        // 4 MB
    float* AO  = VV  + (size_t)ROWS * HID;        // 4 MB
    int*   NBR = (int*)(AO + (size_t)ROWS * HID); // 16384*16 int = 1 MB
    _Float16* WF = (_Float16*)(NBR + (size_t)ROWS * KNN); // 32 KB

    prep_wfrag<<<1, 256, 0, stream>>>(Wq, Wk, Wv, Wo, WF);
    knn_kernel<<<dim3(NPTS / 256, NSLICE), 256, 0, stream>>>(pos, NBR);
    proj_kernel<<<ROWS / 128, 256, 0, stream>>>(x, pos, bq, bk, bv, Wpe, bpe, Wa, WF,
                                                QW, KPE, VV);
    attn_kernel<<<ROWS / 256, 256, 0, stream>>>(QW, KPE, VV, NBR, pos, Wpd, bpd, ba, AO);
    out_kernel<<<ROWS / 128, 256, 0, stream>>>(AO, x, bo, WF, out);
}
